// SSM_31516470018463
// MI455X (gfx1250) — matrix operand order repaired and checked by reading, not compiled, not run on hardware
//
#include <hip/hip_runtime.h>
#include <math.h>

// ---------------------------------------------------------------------------
// Types for CDNA5 WMMA (wave32): v_wmma_f32_16x16x32_bf16
// ---------------------------------------------------------------------------
typedef __attribute__((ext_vector_type(16))) __bf16 v16bf;
typedef __attribute__((ext_vector_type(8)))  __bf16 v8bf;
typedef __attribute__((ext_vector_type(8)))  float  v8f;

#define LANES_C 4
#define HD      256
#define CDIM    1024
#define TDIM    2048
#define BDIM    4
#define BT      (BDIM * TDIM)          // 8192
#define BTC     ((long)BT * CDIM)      // 8,388,608
#define CC      ((long)CDIM * CDIM)    // 1,048,576

// round-to-nearest-even f32 -> bf16 bit pattern
__device__ __forceinline__ unsigned short f2bf(float f) {
  unsigned int u = __float_as_uint(f);
  unsigned int r = u + 0x7FFFu + ((u >> 16) & 1u);
  return (unsigned short)(r >> 16);
}

// ---------------------------------------------------------------------------
// WMMA fragment loads for NT GEMM: C[M,N] = A[M,K] * B[N,K]^T, bf16 row-major
// A 16x32 frag: lane<16 -> row=lane,  K 0..7 (elts 0..7) and 16..23 (elts 8..15)
//               lane>=16 -> same row, K 8..15 and 24..31
// B 32x16 frag: lane<16 -> col=lane,  K 0..7 (elts 0..7) and 16..23 (elts 8..15)
//               lane>=16 -> col=lane-16, K 8..15 and 24..31
// ---------------------------------------------------------------------------
__device__ __forceinline__ v16bf load_a_frag(const unsigned short* __restrict__ A,
                                             int ldk, int rowBase, int k, int lane) {
  int row   = rowBase + (lane & 15);
  int khalf = (lane >> 4) * 8;
  const v8bf* p0 = (const v8bf*)(A + (long)row * ldk + k + khalf);
  const v8bf* p1 = (const v8bf*)(A + (long)row * ldk + k + 16 + khalf);
  v8bf lo = *p0;
  v8bf hi = *p1;
  v16bf out;
#pragma unroll
  for (int i = 0; i < 8; ++i) { out[i] = lo[i]; out[i + 8] = hi[i]; }
  return out;
}

__device__ __forceinline__ v16bf load_b_frag(const unsigned short* __restrict__ B,
                                             int ldk, int colBase, int k, int lane) {
  int col  = colBase + (lane & 15);
  int koff = (lane >> 4) * 8;
  v8bf lo = *(const v8bf*)(B + (long)col * ldk + k + koff);
  v8bf hi = *(const v8bf*)(B + (long)col * ldk + k + 16 + koff);
  v16bf out;
#pragma unroll
  for (int i = 0; i < 8; ++i) { out[i] = lo[i]; out[i + 8] = hi[i]; }
  return out;
}

// ---------------------------------------------------------------------------
// NT GEMM: C = alpha * (A * B^T) + beta * aux   (aux may be null)
// Block: 256 threads = 8 waves (4x2), tile 128(M) x 64(N), K step 32.
// Requires M%128==0, N%64==0, K%32==0 (true for all calls here).
// ---------------------------------------------------------------------------
__global__ __launch_bounds__(256) void gemm_nt_bf16_kernel(
    const unsigned short* __restrict__ A, const unsigned short* __restrict__ B,
    float* __restrict__ C, const float* __restrict__ aux,
    int M, int N, int K, float alpha, float beta) {
  int lane = threadIdx.x & 31;
  int wave = threadIdx.x >> 5;
  int wm = wave >> 1;                 // 0..3
  int wn = wave & 1;                  // 0..1
  int mBase = blockIdx.y * 128 + wm * 32;
  int nBase = blockIdx.x * 64 + wn * 32;

  v8f acc[2][2];
#pragma unroll
  for (int i = 0; i < 2; ++i)
#pragma unroll
    for (int j = 0; j < 2; ++j) acc[i][j] = (v8f){0.f,0.f,0.f,0.f,0.f,0.f,0.f,0.f};

  for (int k = 0; k < K; k += 32) {
    v16bf a0 = load_a_frag(A, K, mBase,      k, lane);
    v16bf a1 = load_a_frag(A, K, mBase + 16, k, lane);
    v16bf b0 = load_b_frag(B, K, nBase,      k, lane);
    v16bf b1 = load_b_frag(B, K, nBase + 16, k, lane);
    acc[0][0] = __builtin_amdgcn_wmma_f32_16x16x32_bf16(false, a0, false, b0,
                                                        (short)0, acc[0][0], false, false);
    acc[0][1] = __builtin_amdgcn_wmma_f32_16x16x32_bf16(false, a0, false, b1,
                                                        (short)0, acc[0][1], false, false);
    acc[1][0] = __builtin_amdgcn_wmma_f32_16x16x32_bf16(false, a1, false, b0,
                                                        (short)0, acc[1][0], false, false);
    acc[1][1] = __builtin_amdgcn_wmma_f32_16x16x32_bf16(false, a1, false, b1,
                                                        (short)0, acc[1][1], false, false);
  }

  // C/D layout: VGPR r: lane<16 -> (M = r,   N = lane), lane>=16 -> (M = r+8, N = lane-16)
  int nCol0 = (lane & 15);
  int rAdd  = (lane >> 4) * 8;
#pragma unroll
  for (int fi = 0; fi < 2; ++fi) {
#pragma unroll
    for (int fj = 0; fj < 2; ++fj) {
      int col = nBase + fj * 16 + nCol0;
#pragma unroll
      for (int r = 0; r < 8; ++r) {
        int row = mBase + fi * 16 + r + rAdd;
        long idx = (long)row * N + col;
        float v = alpha * acc[fi][fj][r];
        if (aux) v += beta * aux[idx];
        C[idx] = v;
      }
    }
  }
}

// ---------------------------------------------------------------------------
// Elementwise helpers
// ---------------------------------------------------------------------------
__global__ void f32_to_bf16_kernel(const float* __restrict__ in,
                                   unsigned short* __restrict__ out, int n) {
  int i = blockIdx.x * blockDim.x + threadIdx.x;
  if (i < n) out[i] = f2bf(in[i]);
}

__global__ void mul_to_bf16_kernel(const float* __restrict__ a,
                                   const float* __restrict__ b,
                                   unsigned short* __restrict__ out, int n) {
  int i = blockIdx.x * blockDim.x + threadIdx.x;
  if (i < n) out[i] = f2bf(a[i] * b[i]);
}

// out_bf[j*dim+i] = bf16(in[i*dim+j])
__global__ void transpose_to_bf16_kernel(const float* __restrict__ in,
                                         unsigned short* __restrict__ out, int dim) {
  int i = blockIdx.x * blockDim.x + threadIdx.x;
  if (i < dim * dim) {
    int r = i / dim, c = i % dim;
    out[(long)c * dim + r] = f2bf(in[i]);
  }
}

// S = bf16(left + right); left[b,t,c] = (t<d) ? identity[c] : qcur[b,t-d,c]
__global__ void shift_sum_bf16_kernel(const float* __restrict__ qcur,
                                      const float* __restrict__ identity,
                                      unsigned short* __restrict__ S, int d) {
  long i = (long)blockIdx.x * blockDim.x + threadIdx.x;
  if (i >= BTC) return;
  int c = (int)(i & (CDIM - 1));
  long token = i >> 10;                 // CDIM == 1024
  int t = (int)(token & (TDIM - 1));
  float left = (t < d) ? identity[c] : qcur[i - (long)d * CDIM];
  S[i] = f2bf(left + qcur[i]);
}

// ---------------------------------------------------------------------------
// Per-token combine: tiny 4x4 SDPA over lanes + RMS norm + residual
// 4 tokens per block (128 threads, 1 wave/token), wave32 shfl reductions.
//   q = left (shifted), k = right (qcur), v = vmix (precomputed GEMM)
// ---------------------------------------------------------------------------
__global__ __launch_bounds__(128) void combine_kernel(
    const float* __restrict__ qcur, const float* __restrict__ vmix,
    const float* __restrict__ identity, float* __restrict__ qnext, int d) {
  __shared__ float kv[4][2 * CDIM];     // [token][ k:0..1023 | v:1024..2047 ]
  int wave = threadIdx.x >> 5;
  int lane = threadIdx.x & 31;
  long token = (long)blockIdx.x * 4 + wave;
  int t = (int)(token & (TDIM - 1));

  const float* right = qcur + token * CDIM;
  const float* vrow  = vmix + token * CDIM;
  const float* leftp = (t < d) ? identity : (qcur + (token - d) * CDIM);

  int l   = lane >> 3;                  // SDPA lane 0..3
  int sub = lane & 7;                   // 8 threads per SDPA lane
  int base = l * HD + sub * 32;         // 32 contiguous channels per thread

  float qreg[32];
#pragma unroll
  for (int j = 0; j < 32; ++j) {
    kv[wave][base + j]        = right[base + j];
    kv[wave][CDIM + base + j] = vrow[base + j];
    qreg[j] = leftp[base + j];
  }
  __syncthreads();

  // scores[l][m] = sum_dd q[l][dd] * k[m][dd] * hd^-0.5
  float sc[LANES_C];
#pragma unroll
  for (int m = 0; m < LANES_C; ++m) {
    float s = 0.f;
    const float* krow = &kv[wave][m * HD + sub * 32];
#pragma unroll
    for (int j = 0; j < 32; ++j) s += qreg[j] * krow[j];
#pragma unroll
    for (int off = 1; off < 8; off <<= 1) s += __shfl_xor(s, off, 32);
    sc[m] = s * 0.0625f;                // 1/sqrt(256)
  }

  // softmax over m
  float mx = fmaxf(fmaxf(sc[0], sc[1]), fmaxf(sc[2], sc[3]));
  float e0 = __expf(sc[0] - mx), e1 = __expf(sc[1] - mx);
  float e2 = __expf(sc[2] - mx), e3 = __expf(sc[3] - mx);
  float inv = 1.f / (e0 + e1 + e2 + e3);
  float att[LANES_C] = {e0 * inv, e1 * inv, e2 * inv, e3 * inv};

  // z = att @ v ; rms over hd ; out = left + z * rsqrt(mean+eps)/(1+l)
  float z[32];
  float ss = 0.f;
#pragma unroll
  for (int j = 0; j < 32; ++j) {
    float zz = 0.f;
#pragma unroll
    for (int m = 0; m < LANES_C; ++m)
      zz += att[m] * kv[wave][CDIM + m * HD + sub * 32 + j];
    z[j] = zz;
    ss += zz * zz;
  }
#pragma unroll
  for (int off = 1; off < 8; off <<= 1) ss += __shfl_xor(ss, off, 32);
  float r = rsqrtf(ss * (1.f / HD) + 1e-8f);
  float scale = r / (float)(1 + l);

  float* out = qnext + token * CDIM + base;
#pragma unroll
  for (int j = 0; j < 32; ++j) out[j] = qreg[j] + z[j] * scale;
}

// ---------------------------------------------------------------------------
// Host orchestration
// ---------------------------------------------------------------------------
static inline void launch_gemm(const unsigned short* A, const unsigned short* B,
                               float* C, const float* aux, int M, int N, int K,
                               float alpha, float beta, hipStream_t s) {
  dim3 grid(N / 64, M / 128);
  gemm_nt_bf16_kernel<<<grid, 256, 0, s>>>(A, B, C, aux, M, N, K, alpha, beta);
}

extern "C" void kernel_launch(void* const* d_in, const int* in_sizes, int n_in,
                              void* d_out, int out_size, void* d_ws, size_t ws_size,
                              hipStream_t stream) {
  const float* x        = (const float*)d_in[0];   // [B,T,C]
  const float* W_up     = (const float*)d_in[1];   // [C,C]
  const float* W_v      = (const float*)d_in[2];   // [C,C]
  const float* W_proj   = (const float*)d_in[3];   // [C,C]
  const float* identity = (const float*)d_in[4];   // [C]
  const float* W_scan   = (const float*)d_in[5];   // [C,C]

  float* Y_out   = (float*)d_out;                  // [B,T,C]
  float* q_final = (float*)d_out + BTC;            // [B,T,C]

  // --- workspace carve-up (all sizes multiples of 4 bytes) ---
  char* w = (char*)d_ws;
  float* qB            = (float*)w;            w += BTC * 4;   // scan ping buffer
  float* v_buf         = (float*)w;            w += BTC * 4;
  float* vmix          = (float*)w;            w += BTC * 4;
  unsigned short* Sbf  = (unsigned short*)w;   w += BTC * 2;
  unsigned short* Xbf  = (unsigned short*)w;   w += BTC * 2;   // reused as P=(q*v) bf16
  float* Y0            = (float*)w;            w += CC * 4;
  float* Y1            = (float*)w;            w += CC * 4;
  float* G             = (float*)w;            w += CC * 4;
  unsigned short* Ybf  = (unsigned short*)w;   w += CC * 2;
  unsigned short* Ytbf = (unsigned short*)w;   w += CC * 2;
  unsigned short* Gbf  = (unsigned short*)w;   w += CC * 2;
  unsigned short* Wupb = (unsigned short*)w;   w += CC * 2;
  unsigned short* Wvb  = (unsigned short*)w;   w += CC * 2;
  unsigned short* Wpb  = (unsigned short*)w;   w += CC * 2;
  unsigned short* Wtb  = (unsigned short*)w;   w += CC * 2;    // (polar W)^T bf16

  const int TPB = 256;
  const int nBTC = (int)BTC, nCC = (int)CC;

  // 1) convert inputs to bf16
  f32_to_bf16_kernel<<<(nBTC + TPB - 1) / TPB, TPB, 0, stream>>>(x, Xbf, nBTC);
  f32_to_bf16_kernel<<<(nCC + TPB - 1) / TPB, TPB, 0, stream>>>(W_up, Wupb, nCC);
  f32_to_bf16_kernel<<<(nCC + TPB - 1) / TPB, TPB, 0, stream>>>(W_v, Wvb, nCC);
  f32_to_bf16_kernel<<<(nCC + TPB - 1) / TPB, TPB, 0, stream>>>(W_proj, Wpb, nCC);

  // 2) q = x @ W_up^T ; v = x @ W_v^T   (both NT GEMMs)
  launch_gemm(Xbf, Wupb, qB, nullptr, BT, CDIM, CDIM, 1.f, 0.f, stream);
  launch_gemm(Xbf, Wvb, v_buf, nullptr, BT, CDIM, CDIM, 1.f, 0.f, stream);

  // 3) Newton-Schulz polar factor: W = UV^T of W_scan_raw
  //    sigma_max ~ 0.02*(2*sqrt(1024)) ~ 1.28 < sqrt(3) -> converges unscaled.
  hipMemcpyAsync(Y0, W_scan, CC * 4, hipMemcpyDeviceToDevice, stream);
  float* cur = Y0; float* nxt = Y1;
  for (int it = 0; it < 30; ++it) {
    f32_to_bf16_kernel<<<(nCC + TPB - 1) / TPB, TPB, 0, stream>>>(cur, Ybf, nCC);
    transpose_to_bf16_kernel<<<(nCC + TPB - 1) / TPB, TPB, 0, stream>>>(cur, Ytbf, CDIM);
    // G = Y^T Y = Yt * Yt^T
    launch_gemm(Ytbf, Ytbf, G, nullptr, CDIM, CDIM, CDIM, 1.f, 0.f, stream);
    f32_to_bf16_kernel<<<(nCC + TPB - 1) / TPB, TPB, 0, stream>>>(G, Gbf, nCC);
    // Ynext = 1.5*Y - 0.5*(Y*G)   (G symmetric -> Y*G = Y * G^T, NT form)
    launch_gemm(Ybf, Gbf, nxt, cur, CDIM, CDIM, CDIM, -0.5f, 1.5f, stream);
    float* tmp = cur; cur = nxt; nxt = tmp;
  }
  // Wt = (polar W)^T in bf16 for the scan GEMM (S @ W = S * Wt^T)
  transpose_to_bf16_kernel<<<(nCC + TPB - 1) / TPB, TPB, 0, stream>>>(cur, Wtb, CDIM);

  // 4) parallel scan: 11 Hillis-Steele combine steps, q ping-pongs qB <-> q_final
  float* qc = qB; float* qn = q_final;
  for (int d = 1; d < TDIM; d <<= 1) {
    shift_sum_bf16_kernel<<<(nBTC + TPB - 1) / TPB, TPB, 0, stream>>>(qc, identity, Sbf, d);
    launch_gemm(Sbf, Wtb, vmix, nullptr, BT, CDIM, CDIM, 1.f, 0.f, stream);
    combine_kernel<<<BT / 4, 128, 0, stream>>>(qc, vmix, identity, qn, d);
    float* tmp = qc; qc = qn; qn = tmp;
  }
  // 11 (odd) swaps: qc == q_final here.

  // 5) Y = (q * v) @ W_proj^T
  mul_to_bf16_kernel<<<(nBTC + TPB - 1) / TPB, TPB, 0, stream>>>(q_final, v_buf, Xbf, nBTC);
  launch_gemm(Xbf, Wpb, Y_out, nullptr, BT, CDIM, CDIM, 1.f, 0.f, stream);
}
